// DressedQuantumNet_36249523978580
// MI455X (gfx1250) — matrix operand order repaired and checked by reading, not compiled, not run on hardware
//
#include <hip/hip_runtime.h>

typedef _Float16 half_t;
typedef __attribute__((ext_vector_type(8)))  _Float16 v8h;
typedef __attribute__((ext_vector_type(16))) _Float16 v16h;
typedef __attribute__((ext_vector_type(8)))  float    v8f;

#define N_ROWS  32768   // 4096 samples * 8 chunks
#define K_DIM   256     // embedded amplitudes per circuit
#define N_DIM   512     // 2^9 output probabilities per circuit
#define N_GATES 114

// Gate table: kind 0=ry 1=rx 2=xx 3=xy 4=cry 5=crx ; param index == gate ordinal.
__device__ __constant__ signed char g_kind[N_GATES] = {
  0,0,0,0,0,0,0,0,0,
  1,1,1,1,1,1,1,1,1,
  2,2,2,2,2,2,2,2,
  4,4,4,4,4,4,4,4,
  0,0,0,0,0,0,0,0,0,
  3,3,3,3,3,3,3,3,
  5,5,5,5,5,5,
  1,1,1,1,1,1,1,1,1,
  2,2,2,2,2,2,2,2,
  4,4,4,4,4,4,4,4,
  0,0,0,0,0,0,0,0,0,
  3,3,3,3,3,3,3,3,
  5,5,5,5,5,5,
  1,1,1,1,1,1,1,1,1
};
__device__ __constant__ signed char g_w0[N_GATES] = {
  0,1,2,3,4,5,6,7,8,
  0,1,2,3,4,5,6,7,8,
  0,8,1,7,2,6,3,5,
  0,1,2,8,7,6,5,3,
  0,1,2,3,4,5,6,7,8,
  4,4,3,5,2,6,1,7,
  0,2,5,7,1,6,
  0,1,2,3,4,5,6,7,8,
  0,8,1,7,2,6,3,5,
  0,1,2,8,7,6,5,3,
  0,1,2,3,4,5,6,7,8,
  4,4,3,5,2,6,1,7,
  0,2,5,7,1,6,
  0,1,2,3,4,5,6,7,8
};
__device__ __constant__ signed char g_w1[N_GATES] = {
  0,0,0,0,0,0,0,0,0,
  0,0,0,0,0,0,0,0,0,
  1,7,2,6,3,5,4,4,
  1,2,3,7,6,5,4,4,
  0,0,0,0,0,0,0,0,0,
  3,5,2,6,1,7,0,8,
  1,3,6,8,2,7,
  0,0,0,0,0,0,0,0,0,
  1,7,2,6,3,5,4,4,
  1,2,3,7,6,5,4,4,
  0,0,0,0,0,0,0,0,0,
  3,5,2,6,1,7,0,8,
  1,3,6,8,2,7,
  0,0,0,0,0,0,0,0,0
};

// ---------------------------------------------------------------------------
// Kernel 1: X[n,k] = normalize(tanh(feats[n,:]))  (f32 -> f16), one wave/row.
// ---------------------------------------------------------------------------
__global__ void prep_x(const float* __restrict__ feats, half_t* __restrict__ X) {
  const int lane = threadIdx.x & 31;
  const int wave = threadIdx.x >> 5;
  const int row  = blockIdx.x * 8 + wave;          // 0 .. 32767
  const float* p = feats + (size_t)row * K_DIM;

  float v[8];
  float ss = 0.f;
#pragma unroll
  for (int i = 0; i < 8; ++i) {
    float t = tanhf(p[lane + 32 * i]);             // pi/2 scale cancels in norm
    v[i] = t;
    ss += t * t;
  }
#pragma unroll
  for (int off = 16; off > 0; off >>= 1)           // wave32 reduction
    ss += __shfl_xor(ss, off, 32);
  const float inv = rsqrtf(fmaxf(ss, 1e-30f));

  half_t* o = X + (size_t)row * K_DIM;
#pragma unroll
  for (int i = 0; i < 8; ++i)
    o[lane + 32 * i] = (half_t)(v[i] * inv);
}

// ---------------------------------------------------------------------------
// Kernel 2: simulate the fixed circuit on basis column k (one WG per column),
// producing Mr/Mi stored transposed: M*[j*256 + k] = U[j, col(k)]  (f16).
// State lives in LDS; each gate partitions the 512 amps into disjoint
// pairs/quads owned by one thread, so only one barrier per gate is needed.
// ---------------------------------------------------------------------------
__global__ void build_unitary(const float* __restrict__ qp,
                              half_t* __restrict__ Mr, half_t* __restrict__ Mi) {
  __shared__ float sr[512];
  __shared__ float si[512];
  const int k = blockIdx.x;                        // input basis column 0..255
  const int t = threadIdx.x;                       // 0..255
  const int col = ((k >> 4) << 5) | (k & 15);      // qubit-4 bit = 0

  for (int j = t; j < 512; j += 256) { sr[j] = (j == col) ? 1.f : 0.f; si[j] = 0.f; }
  __syncthreads();

  for (int g = 0; g < N_GATES; ++g) {
    const int kind = g_kind[g];
    const float th = qp[g] * 0.5f;
    const float c = cosf(th), s = sinf(th);

    if (kind <= 1) {                               // single-qubit ry / rx
      const int p  = 8 - g_w0[g];
      const int m  = 1 << p;
      const int i0 = ((t >> p) << (p + 1)) | (t & (m - 1));
      const int i1 = i0 | m;
      const float r0 = sr[i0], q0 = si[i0], r1 = sr[i1], q1 = si[i1];
      if (kind == 0) {                             // ry: [[c,-s],[s,c]] (real)
        sr[i0] = c * r0 - s * r1;  si[i0] = c * q0 - s * q1;
        sr[i1] = s * r0 + c * r1;  si[i1] = s * q0 + c * q1;
      } else {                                     // rx: [[c,-is],[-is,c]]
        sr[i0] = c * r0 + s * q1;  si[i0] = c * q0 - s * r1;
        sr[i1] = c * r1 + s * q0;  si[i1] = c * q1 - s * r0;
      }
    } else if (t < 128) {                          // two-qubit gates
      const int p0 = 8 - g_w0[g];
      const int p1 = 8 - g_w1[g];
      const int ph = p0 > p1 ? p0 : p1;
      const int pl = p0 > p1 ? p1 : p0;
      int i = t;
      i = ((i >> pl) << (pl + 1)) | (i & ((1 << pl) - 1));
      i = ((i >> ph) << (ph + 1)) | (i & ((1 << ph) - 1));
      const int i00 = i;
      const int i01 = i | (1 << p1);
      const int i10 = i | (1 << p0);
      const int i11 = i | (1 << p0) | (1 << p1);
      if (kind == 2) {                             // xx = exp(-i t/2 XX)
        const float r0 = sr[i00], q0 = si[i00], r3 = sr[i11], q3 = si[i11];
        const float r1 = sr[i01], q1 = si[i01], r2 = sr[i10], q2 = si[i10];
        sr[i00] = c * r0 + s * q3;  si[i00] = c * q0 - s * r3;
        sr[i11] = s * q0 + c * r3;  si[i11] = -s * r0 + c * q3;
        sr[i01] = c * r1 + s * q2;  si[i01] = c * q1 - s * r2;
        sr[i10] = s * q1 + c * r2;  si[i10] = -s * r1 + c * q2;
      } else if (kind == 3) {                      // IsingXY: middle block [[c,is],[is,c]]
        const float r1 = sr[i01], q1 = si[i01], r2 = sr[i10], q2 = si[i10];
        sr[i01] = c * r1 - s * q2;  si[i01] = c * q1 + s * r2;
        sr[i10] = c * r2 - s * q1;  si[i10] = c * q2 + s * r1;
      } else if (kind == 4) {                      // cry: ry on (i10,i11)
        const float r2 = sr[i10], q2 = si[i10], r3 = sr[i11], q3 = si[i11];
        sr[i10] = c * r2 - s * r3;  si[i10] = c * q2 - s * q3;
        sr[i11] = s * r2 + c * r3;  si[i11] = s * q2 + c * q3;
      } else {                                     // crx: rx on (i10,i11)
        const float r2 = sr[i10], q2 = si[i10], r3 = sr[i11], q3 = si[i11];
        sr[i10] = c * r2 + s * q3;  si[i10] = c * q2 - s * r3;
        sr[i11] = s * q2 + c * r3;  si[i11] = -s * r2 + c * q3;
      }
    }
    __syncthreads();
  }

  for (int j = t; j < 512; j += 256) {
    Mr[(size_t)j * K_DIM + k] = (half_t)sr[j];
    Mi[(size_t)j * K_DIM + k] = (half_t)si[j];
  }
}

// ---------------------------------------------------------------------------
// Kernel 3: probs = (X @ Mr^T)^2 + (X @ Mi^T)^2 via v_wmma_f32_16x16x32_f16.
// One wave per 16x16 tile; 8 waves/block share the A (row) tile.
// ---------------------------------------------------------------------------
__global__ void wmma_gemm(const half_t* __restrict__ X,
                          const half_t* __restrict__ Mr,
                          const half_t* __restrict__ Mi,
                          float* __restrict__ out) {
  const int lane  = threadIdx.x & 31;
  const int wave  = threadIdx.x >> 5;
  const int mtile = blockIdx.x;                    // 0..2047
  const int ntile = blockIdx.y * 8 + wave;         // 0..31
  const int rc    = lane & 15;                     // A row / B column in tile
  const int hi    = lane >> 4;                     // lane half select
  const int khA   = hi << 4;                       // A: +0 or +16 (contiguous K, matches B)
  const int khB   = hi << 4;                       // B: +0 or +16 (contiguous K, matches A)

  const half_t* ap = X  + (size_t)(mtile * 16 + rc) * K_DIM;
  const half_t* rp = Mr + (size_t)(ntile * 16 + rc) * K_DIM;
  const half_t* ip = Mi + (size_t)(ntile * 16 + rc) * K_DIM;

  v8f accr = {};
  v8f acci = {};
#pragma unroll
  for (int kt = 0; kt < K_DIM; kt += 32) {
    // A fragment: lane<16 holds K 0..15, lane>=16 holds K 16..31
    v8h a0 = *(const v8h*)(ap + kt + khA);
    v8h a1 = *(const v8h*)(ap + kt + khA + 8);
    v16h a = __builtin_shufflevector(a0, a1, 0,1,2,3,4,5,6,7,8,9,10,11,12,13,14,15);
    // B fragment: lane<16 holds K 0..15, lane>=16 holds K 16..31 (contiguous
    // thanks to transposed M storage)
    v8h b0 = *(const v8h*)(rp + kt + khB);
    v8h b1 = *(const v8h*)(rp + kt + khB + 8);
    v16h br = __builtin_shufflevector(b0, b1, 0,1,2,3,4,5,6,7,8,9,10,11,12,13,14,15);
    v8h c0 = *(const v8h*)(ip + kt + khB);
    v8h c1 = *(const v8h*)(ip + kt + khB + 8);
    v16h bi = __builtin_shufflevector(c0, c1, 0,1,2,3,4,5,6,7,8,9,10,11,12,13,14,15);

    accr = __builtin_amdgcn_wmma_f32_16x16x32_f16(false, a, false, br, (short)0, accr, false, false);
    acci = __builtin_amdgcn_wmma_f32_16x16x32_f16(false, a, false, bi, (short)0, acci, false, false);
  }

  // D layout (f32 16x16): lane holds column n = lane&15, rows m = hi*8 + r.
  const int n = rc;
  const int mbase = hi << 3;
  float* op = out + (size_t)(mtile * 16 + mbase) * N_DIM + ntile * 16 + n;
#pragma unroll
  for (int r = 0; r < 8; ++r)
    op[(size_t)r * N_DIM] = accr[r] * accr[r] + acci[r] * acci[r];
}

// ---------------------------------------------------------------------------
extern "C" void kernel_launch(void* const* d_in, const int* in_sizes, int n_in,
                              void* d_out, int out_size, void* d_ws, size_t ws_size,
                              hipStream_t stream) {
  const float* feats = (const float*)d_in[0];      // (4096, 2048) f32
  const float* qp    = (const float*)d_in[1];      // (114,) f32
  float* out = (float*)d_out;                      // (4096,1024,2,2) == (32768,512)

  char* ws = (char*)d_ws;
  half_t* X  = (half_t*)ws;                                      // 16 MB
  half_t* Mr = (half_t*)(ws + (size_t)N_ROWS * K_DIM * sizeof(half_t));
  half_t* Mi = Mr + (size_t)N_DIM * K_DIM;                       // 256 KB each

  prep_x<<<N_ROWS / 8, 256, 0, stream>>>(feats, X);
  build_unitary<<<K_DIM, 256, 0, stream>>>(qp, Mr, Mi);
  wmma_gemm<<<dim3(N_ROWS / 16, 4), 256, 0, stream>>>(X, Mr, Mi, out);
}
